// RingDilatedAttentionHilbertCoreFixed_65755949301797
// MI455X (gfx1250) — matrix operand order repaired and checked by reading, not compiled, not run on hardware
//
#include <hip/hip_runtime.h>

typedef _Float16 f16;
typedef _Float16 v16h __attribute__((ext_vector_type(16)));
typedef _Float16 v8h  __attribute__((ext_vector_type(8)));
typedef float    v8f  __attribute__((ext_vector_type(8)));
typedef float    v4f  __attribute__((ext_vector_type(4)));

#define SEQ   14336
#define DIMD  1024
#define HEADS 16
#define HD    64
#define LRR   2048   // L/rate is 2048 for every segment
#define NRT32 448    // SEQ/32 row tiles

__device__ __forceinline__ int d_seg_of_row(int row) { return row < 2048 ? 0 : (row < 6144 ? 1 : 2); }
__device__ __forceinline__ int d_seg_pos(int s)      { return s == 0 ? 0 : (s == 1 ? 2048 : 6144); }
__device__ __forceinline__ int d_seg_len(int s)      { return s == 0 ? 2048 : (s == 1 ? 4096 : 8192); }

__device__ __forceinline__ v8f wmma16(v16h a, v16h b, v8f c) {
  return __builtin_amdgcn_wmma_f32_16x16x32_f16(false, a, false, b, (short)0, c, false, false);
}

__device__ __forceinline__ v16h ld_chunks(const f16* p0, const f16* p1) {
  v8h lo = *(const v8h*)p0;
  v8h hi = *(const v8h*)p1;
  v16h r;
#pragma unroll
  for (int i = 0; i < 8; ++i) { r[i] = lo[i]; r[8 + i] = hi[i]; }
  return r;
}
// A operand (16-bit 16x32): lane row m=lane&15, g=lane>>4; contiguous K {kofs+16g..+15}
__device__ __forceinline__ v16h ld_a(const f16* row, int kofs, int g) {
  return ld_chunks(row + kofs + 16 * g, row + kofs + 16 * g + 8);
}
// B operand (16-bit 32x16): lane col n=lane&15; contiguous 16 K-values (caller adds 16*g)
__device__ __forceinline__ v16h ld_b(const f16* p) { return ld_chunks(p, p + 8); }

// A operand built from fp32 source (convert to f16 in-register)
__device__ __forceinline__ v16h ld_a_f32(const float* px, int kk, int g) {
  const float* p0 = px + kk + 16 * g;
  const float* p1 = px + kk + 16 * g + 8;
  v4f a0 = *(const v4f*)p0;
  v4f a1 = *(const v4f*)(p0 + 4);
  v4f a2 = *(const v4f*)p1;
  v4f a3 = *(const v4f*)(p1 + 4);
  v16h a;
#pragma unroll
  for (int i = 0; i < 4; ++i) {
    a[i] = (f16)a0[i]; a[4 + i] = (f16)a1[i];
    a[8 + i] = (f16)a2[i]; a[12 + i] = (f16)a3[i];
  }
  return a;
}

// ---------------- Hilbert permutation on device ----------------
__global__ __launch_bounds__(256) void k_hilbert_d(int* __restrict__ dbuf) {
  int i = blockIdx.x * blockDim.x + threadIdx.x;
  if (i >= SEQ) return;
  int seg = d_seg_of_row(i);
  int il = i - d_seg_pos(seg);
  int logn = (seg == 2) ? 7 : 6;        // n=128 for L=8192, else n=64
  int n = 1 << logn;
  int x = il & (n - 1);
  int y = il >> logn;
  int d = 0;
  for (int s = n >> 1; s > 0; s >>= 1) {
    int rx = (x & s) ? 1 : 0;
    int ry = (y & s) ? 1 : 0;
    d += s * s * ((3 * rx) ^ ry);
    if (ry == 0) {
      if (rx == 1) { x = s - 1 - x; y = s - 1 - y; }
      int t = x; x = y; y = t;
    }
  }
  dbuf[i] = d;
}

__global__ __launch_bounds__(256) void k_rank(const int* __restrict__ dbuf, int* __restrict__ perm) {
  int i = blockIdx.x * blockDim.x + threadIdx.x;
  if (i >= SEQ) return;
  int seg = d_seg_of_row(i);
  int pos = d_seg_pos(seg);
  int L = d_seg_len(seg);
  int il = i - pos;
  int di = dbuf[i];
  const int* db = dbuf + pos;
  int rank = 0;
  for (int j = 0; j < L; ++j) rank += (db[j] < di) ? 1 : 0;
  perm[pos + rank] = il;   // hilbert distances distinct -> race-free scatter
}

// ---------------- weight convert + transpose (fp32 [K][N] -> f16 [N][K]) ----------------
__global__ __launch_bounds__(256) void k_cvt_wqkv(const float* __restrict__ w0, const float* __restrict__ w1,
                                                  const float* __restrict__ w2, f16* __restrict__ wT) {
  const size_t total = (size_t)3 * 3072 * 1024;
  for (size_t i = blockIdx.x * (size_t)blockDim.x + threadIdx.x; i < total;
       i += (size_t)gridDim.x * blockDim.x) {
    int seg = (int)(i / (3072u * 1024u));
    int rem = (int)(i % (3072u * 1024u));
    int n = rem >> 10, k = rem & 1023;
    const float* w = seg == 0 ? w0 : (seg == 1 ? w1 : w2);
    wT[i] = (f16)w[(size_t)k * 3072 + n];
  }
}

__global__ __launch_bounds__(256) void k_cvt_wo(const float* __restrict__ w0, const float* __restrict__ w1,
                                                const float* __restrict__ w2, f16* __restrict__ wT) {
  const size_t total = (size_t)3 * 1024 * 1024;
  for (size_t i = blockIdx.x * (size_t)blockDim.x + threadIdx.x; i < total;
       i += (size_t)gridDim.x * blockDim.x) {
    int seg = (int)(i >> 20);
    int rem = (int)(i & 1048575u);
    int n = rem >> 10, k = rem & 1023;
    const float* w = seg == 0 ? w0 : (seg == 1 ? w1 : w2);
    wT[i] = (f16)w[(size_t)k * 1024 + n];
  }
}

__global__ __launch_bounds__(256) void k_zero(uint4* __restrict__ p, size_t n4) {
  uint4 z; z.x = z.y = z.z = z.w = 0u;
  for (size_t i = blockIdx.x * (size_t)blockDim.x + threadIdx.x; i < n4;
       i += (size_t)gridDim.x * blockDim.x) p[i] = z;
}

// ---------------- GEMM1: xh(f32,perm-gather) @ wqkvT -> dilated Qd/Kd/VdT (f16) ----------------
// 32(M) x 64(N) per wave, double-buffered + unroll-2 (buffer role swap, no register copies).
__global__ __launch_bounds__(256) void k_gemm_qkv(
    const float* __restrict__ x, const int* __restrict__ perm, const f16* __restrict__ wT,
    const float* __restrict__ bq0, const float* __restrict__ bq1, const float* __restrict__ bq2,
    f16* __restrict__ Qd, f16* __restrict__ Kd, f16* __restrict__ VdT) {
  int wid = (blockIdx.x * blockDim.x + threadIdx.x) >> 5;
  int lane = threadIdx.x & 31;
  int rt = wid % NRT32;
  int nt64 = wid / NRT32;                // 0..47
  if (nt64 >= 48) return;
  int seg = rt < 64 ? 0 : (rt < 192 ? 1 : 2);
  int segpos = d_seg_pos(seg);
  int m = lane & 15, g = lane >> 4;

  int pr0 = perm[rt * 32 + m];           // hilbert gather rows (segment-local)
  int pr1 = perm[rt * 32 + 16 + m];
  const float* px0 = x + (size_t)(segpos + pr0) * DIMD;
  const float* px1 = x + (size_t)(segpos + pr1) * DIMD;
  const f16* wseg = wT + (size_t)seg * 3072 * 1024;
  const float* bias = seg == 0 ? bq0 : (seg == 1 ? bq1 : bq2);
  int nb = nt64 * 64;
  const f16* wb0 = wseg + (size_t)(nb + 0 * 16 + m) * DIMD + 16 * g;
  const f16* wb1 = wseg + (size_t)(nb + 1 * 16 + m) * DIMD + 16 * g;
  const f16* wb2 = wseg + (size_t)(nb + 2 * 16 + m) * DIMD + 16 * g;
  const f16* wb3 = wseg + (size_t)(nb + 3 * 16 + m) * DIMD + 16 * g;

  v8f c0[4], c1[4];
#pragma unroll
  for (int t = 0; t < 4; ++t)
#pragma unroll
    for (int i = 0; i < 8; ++i) { c0[t][i] = 0.f; c1[t][i] = 0.f; }

#define GSTEP(AA0, AA1, BB0, BB1, BB2, BB3)                             \
  c0[0] = wmma16(AA0, BB0, c0[0]); c1[0] = wmma16(AA1, BB0, c1[0]);     \
  c0[1] = wmma16(AA0, BB1, c0[1]); c1[1] = wmma16(AA1, BB1, c1[1]);     \
  c0[2] = wmma16(AA0, BB2, c0[2]); c1[2] = wmma16(AA1, BB2, c1[2]);     \
  c0[3] = wmma16(AA0, BB3, c0[3]); c1[3] = wmma16(AA1, BB3, c1[3]);

  // prologue: stage 0 into buffer set 0
  v16h a0 = ld_a_f32(px0, 0, g);
  v16h a1 = ld_a_f32(px1, 0, g);
  v16h b0 = ld_b(wb0), b1 = ld_b(wb1), b2 = ld_b(wb2), b3 = ld_b(wb3);
  v16h ua0, ua1, ub0, ub1, ub2, ub3;

#pragma unroll 1
  for (int kk = 0; kk < DIMD - 96; kk += 64) {
    __builtin_prefetch(px0 + kk + 160, 0, 1);
    // load stage kk+32 into buffer set 1, compute stage kk from set 0
    ua0 = ld_a_f32(px0, kk + 32, g);
    ua1 = ld_a_f32(px1, kk + 32, g);
    ub0 = ld_b(wb0 + kk + 32); ub1 = ld_b(wb1 + kk + 32);
    ub2 = ld_b(wb2 + kk + 32); ub3 = ld_b(wb3 + kk + 32);
    GSTEP(a0, a1, b0, b1, b2, b3)
    // load stage kk+64 into buffer set 0, compute stage kk+32 from set 1
    a0 = ld_a_f32(px0, kk + 64, g);
    a1 = ld_a_f32(px1, kk + 64, g);
    b0 = ld_b(wb0 + kk + 64); b1 = ld_b(wb1 + kk + 64);
    b2 = ld_b(wb2 + kk + 64); b3 = ld_b(wb3 + kk + 64);
    GSTEP(ua0, ua1, ub0, ub1, ub2, ub3)
  }
  // tail: set 0 holds stage DIMD-64; load final stage DIMD-32 into set 1
  ua0 = ld_a_f32(px0, DIMD - 32, g);
  ua1 = ld_a_f32(px1, DIMD - 32, g);
  ub0 = ld_b(wb0 + DIMD - 32); ub1 = ld_b(wb1 + DIMD - 32);
  ub2 = ld_b(wb2 + DIMD - 32); ub3 = ld_b(wb3 + DIMD - 32);
  GSTEP(a0, a1, b0, b1, b2, b3)
  GSTEP(ua0, ua1, ub0, ub1, ub2, ub3)

  // epilogue: bias + dilation-scatter into Qd/Kd/VdT
  int rate = 1 << seg;
#pragma unroll
  for (int nt = 0; nt < 4; ++nt) {
    int cg = nb + nt * 16 + m;
    int which = cg >> 10;                // 0=q 1=k 2=v
    int rem = cg & 1023;
    int h = rem >> 6, d = rem & 63;
    float bv = bias[cg];
    int off = h & (rate - 1);
    size_t qb = ((size_t)(seg * 16 + h) * LRR) * 64 + d;       // Qd/Kd [sh][j][d]
    size_t vb = ((size_t)(seg * 16 + h) * 64 + d) * LRR;       // VdT  [sh][d][j]
#pragma unroll
    for (int rr = 0; rr < 2; ++rr) {
#pragma unroll
      for (int r = 0; r < 8; ++r) {
        int aL = rt * 32 + rr * 16 + r + 8 * g - segpos;
        if ((aL & (rate - 1)) == off) {
          int j = aL >> seg;
          float cv = (rr == 0) ? c0[nt][r] : c1[nt][r];
          f16 hv = (f16)(cv + bv);
          if (which == 0)      Qd[qb + (size_t)j * 64] = hv;
          else if (which == 1) Kd[qb + (size_t)j * 64] = hv;
          else                 VdT[vb + j] = hv;
        }
      }
    }
  }
}

// ---------------- attention: flash-style, transposed scores, pipelined K/V loads ----------------
__global__ __launch_bounds__(256) void k_attn(const f16* __restrict__ Qd, const f16* __restrict__ Kd,
                                              const f16* __restrict__ VdT, f16* __restrict__ attn) {
  int wid = (blockIdx.x * blockDim.x + threadIdx.x) >> 5;
  int lane = threadIdx.x & 31;
  if (wid >= 3 * HEADS * (LRR / 16)) return;
  int seg = wid >> 11;                 // 2048 tiles per segment
  int rem = wid & 2047;
  int h = rem >> 7;
  int qt = rem & 127;
  int m = lane & 15, g = lane >> 4;
  int j0 = qt * 16;

  const f16* q_h = Qd + ((size_t)(seg * 16 + h) * LRR) * 64;
  const f16* k_row = Kd + ((size_t)(seg * 16 + h) * LRR) * 64 + (size_t)m * 64;
  const f16* v_h = VdT + ((size_t)(seg * 16 + h) * 64) * LRR;

  // Q as B operand (held for whole key loop): lane = query m, contiguous hd chunk 16g
  v16h bq0 = ld_b(q_h + (size_t)(j0 + m) * 64 + 16 * g);
  v16h bq1 = ld_b(q_h + (size_t)(j0 + m) * 64 + 32 + 16 * g);

  v8f acc[4];
#pragma unroll
  for (int t = 0; t < 4; ++t)
#pragma unroll
    for (int i = 0; i < 8; ++i) acc[t][i] = 0.f;
  float runm = -1e30f, lsum = 0.f;
  const float scale = 0.125f;          // 1/sqrt(64)

  // prologue: K operands for first key block
  v16h ka00 = ld_a(k_row, 0, g);
  v16h ka01 = ld_a(k_row, 32, g);
  v16h ka10 = ld_a(k_row + 16 * 64, 0, g);
  v16h ka11 = ld_a(k_row + 16 * 64, 32, g);

#pragma unroll 1
  for (int kb = 0; kb < LRR; kb += 32) {
    // scores (transposed): 2 key sub-tiles x 2 hd halves
    v8f s0, s1;
#pragma unroll
    for (int i = 0; i < 8; ++i) { s0[i] = 0.f; s1[i] = 0.f; }
    s0 = wmma16(ka00, bq0, s0);
    s0 = wmma16(ka01, bq1, s0);
    s1 = wmma16(ka10, bq0, s1);
    s1 = wmma16(ka11, bq1, s1);

    // issue V loads for this block + K loads for next block (clamped) before softmax VALU
    v16h va0 = ld_a(v_h + (size_t)(0 + m) * LRR + kb, 0, g);
    v16h va1 = ld_a(v_h + (size_t)(16 + m) * LRR + kb, 0, g);
    v16h va2 = ld_a(v_h + (size_t)(32 + m) * LRR + kb, 0, g);
    v16h va3 = ld_a(v_h + (size_t)(48 + m) * LRR + kb, 0, g);
    int kn = (kb + 32 < LRR) ? (kb + 32) : kb;
    __builtin_prefetch(k_row + (size_t)(kn + 32) * 64, 0, 1);
    ka00 = ld_a(k_row + (size_t)kn * 64, 0, g);
    ka01 = ld_a(k_row + (size_t)kn * 64, 32, g);
    ka10 = ld_a(k_row + (size_t)(kn + 16) * 64, 0, g);
    ka11 = ld_a(k_row + (size_t)(kn + 16) * 64, 32, g);

    float p0[8], p1[8];
    float bm = -1e30f;
#pragma unroll
    for (int r = 0; r < 8; ++r) {
      p0[r] = s0[r] * scale; bm = fmaxf(bm, p0[r]);
      p1[r] = s1[r] * scale; bm = fmaxf(bm, p1[r]);
    }
    bm = fmaxf(bm, __shfl_xor(bm, 16, 32));
    float nm = fmaxf(runm, bm);
    float alpha = __expf(runm - nm);
    runm = nm;
    float ps = 0.f;
#pragma unroll
    for (int r = 0; r < 8; ++r) {
      p0[r] = __expf(p0[r] - nm); ps += p0[r];
      p1[r] = __expf(p1[r] - nm); ps += p1[r];
    }
    ps += __shfl_xor(ps, 16, 32);
    lsum = lsum * alpha + ps;
#pragma unroll
    for (int t = 0; t < 4; ++t)
#pragma unroll
      for (int i = 0; i < 8; ++i) acc[t][i] *= alpha;

    // build P^T as B operand: lane<16 needs keys kb..kb+15, lane>=16 needs kb+16..kb+31
    v16h bp;
#pragma unroll
    for (int r = 0; r < 8; ++r) {
      float q0 = __shfl_xor(p0[r], 16, 32);
      float q1 = __shfl_xor(p1[r], 16, 32);
      float lo = (g == 0) ? p0[r] : q1;
      float hi = (g == 0) ? q0 : p1[r];
      bp[r] = (f16)lo;
      bp[8 + r] = (f16)hi;
    }
    acc[0] = wmma16(va0, bp, acc[0]);
    acc[1] = wmma16(va1, bp, acc[1]);
    acc[2] = wmma16(va2, bp, acc[2]);
    acc[3] = wmma16(va3, bp, acc[3]);
  }

  float invl = 1.f / lsum;
  int rate = 1 << seg;
  int off = h & (rate - 1);
  int grow = d_seg_pos(seg) + off + rate * (j0 + m);   // dilated scatter row
  f16* orow = attn + (size_t)grow * DIMD + h * 64;
#pragma unroll
  for (int t = 0; t < 4; ++t) {
    v8h o;
#pragma unroll
    for (int i = 0; i < 8; ++i) o[i] = (f16)(acc[t][i] * invl);
    *(v8h*)(orow + t * 16 + 8 * g) = o;
  }
}

// ---------------- GEMM2: attn(f16) @ woT -> d_out (fp32, inverse-hilbert scatter) ----------------
// 32(M) x 64(N) per wave, double-buffered + unroll-2 (buffer role swap).
__global__ __launch_bounds__(256) void k_gemm_out(
    const f16* __restrict__ attn, const f16* __restrict__ woT, const int* __restrict__ perm,
    const float* __restrict__ bo0, const float* __restrict__ bo1, const float* __restrict__ bo2,
    float* __restrict__ out) {
  int wid = (blockIdx.x * blockDim.x + threadIdx.x) >> 5;
  int lane = threadIdx.x & 31;
  int rt = wid % NRT32;
  int nt64 = wid / NRT32;                // 0..15
  if (nt64 >= 16) return;
  int seg = rt < 64 ? 0 : (rt < 192 ? 1 : 2);
  int segpos = d_seg_pos(seg);
  int m = lane & 15, g = lane >> 4;

  const f16* pa0 = attn + (size_t)(rt * 32 + m) * DIMD;
  const f16* pa1 = attn + (size_t)(rt * 32 + 16 + m) * DIMD;
  const f16* wseg = woT + (size_t)seg * 1024 * 1024;
  const float* bias = seg == 0 ? bo0 : (seg == 1 ? bo1 : bo2);
  int nb = nt64 * 64;
  const f16* wb0 = wseg + (size_t)(nb + 0 * 16 + m) * DIMD + 16 * g;
  const f16* wb1 = wseg + (size_t)(nb + 1 * 16 + m) * DIMD + 16 * g;
  const f16* wb2 = wseg + (size_t)(nb + 2 * 16 + m) * DIMD + 16 * g;
  const f16* wb3 = wseg + (size_t)(nb + 3 * 16 + m) * DIMD + 16 * g;

  v8f c0[4], c1[4];
#pragma unroll
  for (int t = 0; t < 4; ++t)
#pragma unroll
    for (int i = 0; i < 8; ++i) { c0[t][i] = 0.f; c1[t][i] = 0.f; }

  v16h a0 = ld_a(pa0, 0, g);
  v16h a1 = ld_a(pa1, 0, g);
  v16h b0 = ld_b(wb0), b1 = ld_b(wb1), b2 = ld_b(wb2), b3 = ld_b(wb3);
  v16h ua0, ua1, ub0, ub1, ub2, ub3;

#pragma unroll 1
  for (int kk = 0; kk < DIMD - 96; kk += 64) {
    __builtin_prefetch(pa0 + kk + 160, 0, 1);
    ua0 = ld_a(pa0, kk + 32, g);
    ua1 = ld_a(pa1, kk + 32, g);
    ub0 = ld_b(wb0 + kk + 32); ub1 = ld_b(wb1 + kk + 32);
    ub2 = ld_b(wb2 + kk + 32); ub3 = ld_b(wb3 + kk + 32);
    GSTEP(a0, a1, b0, b1, b2, b3)
    a0 = ld_a(pa0, kk + 64, g);
    a1 = ld_a(pa1, kk + 64, g);
    b0 = ld_b(wb0 + kk + 64); b1 = ld_b(wb1 + kk + 64);
    b2 = ld_b(wb2 + kk + 64); b3 = ld_b(wb3 + kk + 64);
    GSTEP(ua0, ua1, ub0, ub1, ub2, ub3)
  }
  ua0 = ld_a(pa0, DIMD - 32, g);
  ua1 = ld_a(pa1, DIMD - 32, g);
  ub0 = ld_b(wb0 + DIMD - 32); ub1 = ld_b(wb1 + DIMD - 32);
  ub2 = ld_b(wb2 + DIMD - 32); ub3 = ld_b(wb3 + DIMD - 32);
  GSTEP(a0, a1, b0, b1, b2, b3)
  GSTEP(ua0, ua1, ub0, ub1, ub2, ub3)

  int prow0[8], prow1[8];
#pragma unroll
  for (int r = 0; r < 8; ++r) {
    prow0[r] = perm[rt * 32 + r + 8 * g];
    prow1[r] = perm[rt * 32 + 16 + r + 8 * g];
  }
#pragma unroll
  for (int nt = 0; nt < 4; ++nt) {
    int cg = nb + nt * 16 + m;
    float bv = bias[cg];
#pragma unroll
    for (int r = 0; r < 8; ++r) {
      out[(size_t)(segpos + prow0[r]) * DIMD + cg] = c0[nt][r] + bv;
      out[(size_t)(segpos + prow1[r]) * DIMD + cg] = c1[nt][r] + bv;
    }
  }
}

extern "C" void kernel_launch(void* const* d_in, const int* in_sizes, int n_in,
                              void* d_out, int out_size, void* d_ws, size_t ws_size,
                              hipStream_t stream) {
  (void)in_sizes; (void)n_in; (void)out_size; (void)ws_size;
  const float* x     = (const float*)d_in[0];
  const float* wqkv0 = (const float*)d_in[3];
  const float* bqkv0 = (const float*)d_in[4];
  const float* wo0   = (const float*)d_in[5];
  const float* bo0   = (const float*)d_in[6];
  const float* wqkv1 = (const float*)d_in[7];
  const float* bqkv1 = (const float*)d_in[8];
  const float* wo1   = (const float*)d_in[9];
  const float* bo1   = (const float*)d_in[10];
  const float* wqkv2 = (const float*)d_in[11];
  const float* bqkv2 = (const float*)d_in[12];
  const float* wo2   = (const float*)d_in[13];
  const float* bo2   = (const float*)d_in[14];

  char* p = (char*)d_ws;
  auto take = [&](size_t bytes) { char* r = p; p += (bytes + 255) & ~(size_t)255; return r; };
  int* dbuf   = (int*)take((size_t)SEQ * 4);
  int* perm   = (int*)take((size_t)SEQ * 4);
  f16* wqkvT  = (f16*)take((size_t)3 * 3072 * 1024 * 2);
  f16* woT    = (f16*)take((size_t)3 * 1024 * 1024 * 2);
  f16* Qd     = (f16*)take((size_t)3 * HEADS * LRR * 64 * 2);
  f16* Kd     = (f16*)take((size_t)3 * HEADS * LRR * 64 * 2);
  f16* VdT    = (f16*)take((size_t)3 * HEADS * LRR * 64 * 2);
  f16* attn   = (f16*)take((size_t)SEQ * DIMD * 2);
  float* out  = (float*)d_out;

  k_hilbert_d<<<SEQ / 256, 256, 0, stream>>>(dbuf);
  k_rank<<<SEQ / 256, 256, 0, stream>>>(dbuf, perm);
  k_cvt_wqkv<<<4096, 256, 0, stream>>>(wqkv0, wqkv1, wqkv2, wqkvT);
  k_cvt_wo<<<2048, 256, 0, stream>>>(wo0, wo1, wo2, woT);
  k_zero<<<2048, 256, 0, stream>>>((uint4*)attn, (size_t)SEQ * DIMD * 2 / 16);
  k_gemm_qkv<<<(NRT32 * 48) / 8, 256, 0, stream>>>(x, perm, wqkvT, bqkv0, bqkv1, bqkv2, Qd, Kd, VdT);
  k_attn<<<(3 * HEADS * (LRR / 16)) / 8, 256, 0, stream>>>(Qd, Kd, VdT, attn);
  k_gemm_out<<<(NRT32 * 16) / 8, 256, 0, stream>>>(attn, woT, perm, bo0, bo1, bo2, out);
}
